// SchNetConv_4380866641943
// MI455X (gfx1250) — matrix operand order repaired and checked by reading, not compiled, not run on hardware
//
#include <hip/hip_runtime.h>
#include <hip/hip_bf16.h>

#define N_NODES 20000
#define N_EDGES 640000
#define HIDDEN  128
#define RBF_DIM 32

typedef __attribute__((ext_vector_type(16))) __bf16         v16bf;
typedef __attribute__((ext_vector_type(8)))  __bf16         v8bf;
typedef __attribute__((ext_vector_type(16))) unsigned short v16u;
typedef __attribute__((ext_vector_type(8)))  float          v8f;

struct Bf16x16 { v8bf lo, hi; };   // two contiguous 16B halves of an A fragment

// ShiftedSoftplus: softplus(v) - 0.5
static __device__ __forceinline__ float ssp(float v) {
  float sp = __logf(1.0f + __expf(v));
  sp = (v > 15.0f) ? v : sp;
  return sp - 0.5f;
}

static __device__ __forceinline__ v8f wmma_bf16(v16bf a, v16bf b, v8f c) {
  return __builtin_amdgcn_wmma_f32_16x16x32_bf16(
      false, a, false, b, (short)0, c, false, false);
}

static __device__ __forceinline__ v16bf load_b_frag(const __bf16* p) {
  return __builtin_bit_cast(v16bf, *(const v16u*)p);   // 32B aligned vector load
}

// ---------------------------------------------------------------------------
// zero aggr + deg every launch (atomics accumulate into them)
__global__ void zero_kernel(float* __restrict__ p, int n) {
  int t = blockIdx.x * blockDim.x + threadIdx.x;
  if (t < n) p[t] = 0.0f;
}

// convert + transpose weights to bf16 [N][K] layout for WMMA B fragments
__global__ void prep_kernel(const float* __restrict__ fw1, const float* __restrict__ fw2,
                            const float* __restrict__ uw1, const float* __restrict__ uw2,
                            __bf16* __restrict__ fw1t, __bf16* __restrict__ fw2t,
                            __bf16* __restrict__ uw1t, __bf16* __restrict__ uw2t) {
  int t = blockIdx.x * blockDim.x + threadIdx.x;
  if (t < 4096) {                       // fw1: [32,128] -> fw1t[n*32+k]
    int n = t >> 5, k = t & 31;
    fw1t[t] = (__bf16)fw1[k * HIDDEN + n];
  } else if (t < 4096 + 16384) {        // fw2: [128,128] -> fw2t[n*128+k]
    int s = t - 4096; int n = s >> 7, k = s & 127;
    fw2t[s] = (__bf16)fw2[k * HIDDEN + n];
  } else if (t < 4096 + 2 * 16384) {
    int s = t - 4096 - 16384; int n = s >> 7, k = s & 127;
    uw1t[s] = (__bf16)uw1[k * HIDDEN + n];
  } else if (t < 4096 + 3 * 16384) {
    int s = t - 4096 - 2 * 16384; int n = s >> 7, k = s & 127;
    uw2t[s] = (__bf16)uw2[k * HIDDEN + n];
  }
}

// ---------------------------------------------------------------------------
// Edge kernel: one wave = 16 edges. RBF -> WMMA GEMM1 -> ssp -> WMMA GEMM2
// -> m_ij = h[j]*W -> atomic scatter-add into aggr[i], degree count.
__global__ __launch_bounds__(256) void edge_kernel(
    const float* __restrict__ h, const float* __restrict__ x,
    const int* __restrict__ ei,                // [2*E]: row0=i(dst), row1=j(src)
    const float* __restrict__ fb1, const float* __restrict__ fb2,
    const __bf16* __restrict__ fw1t,
    const __bf16* __restrict__ fw2t,
    float* __restrict__ aggr, float* __restrict__ deg) {
  __shared__ __bf16 lds[8][16 * HIDDEN];
  const int lane = threadIdx.x & 31;
  const int wave = threadIdx.x >> 5;
  const int m    = lane & 15;          // M index (edge-in-tile / N column)
  const int kh   = lane >> 4;          // lane-half selector
  const int tile = blockIdx.x * 8 + wave;
  const int base = tile * 16;

  // --- distance for edge (base+m); lanes 16..31 duplicate lanes 0..15
  const int e0 = base + m;
  const int i0 = ei[e0];
  const int j0 = ei[N_EDGES + e0];
  const float dx = x[3 * i0 + 0] - x[3 * j0 + 0];
  const float dy = x[3 * i0 + 1] - x[3 * j0 + 1];
  const float dz = x[3 * i0 + 2] - x[3 * j0 + 2];
  const float dist = __fsqrt_rn(dx * dx + dy * dy + dz * dz);

  // --- Gaussian RBF built directly into the 16x32 bf16 A fragment
  v16bf a1;
#pragma unroll
  for (int e = 0; e < 16; ++e) {
    const int k   = e + kh * 16;                          // contiguous A layout
    const float c = (float)k * (10.0f / 31.0f);           // linspace(0,10,32)
    const float d = dist - c;
    a1[e] = (__bf16)__expf(-10.0f * d * d);
  }

  // --- GEMM1: rbf[16x32] @ fw1[32x128]
  const v8f z = {0.f, 0.f, 0.f, 0.f, 0.f, 0.f, 0.f, 0.f};
  v8f acc[8];
#pragma unroll
  for (int nt = 0; nt < 8; ++nt) {
    const v16bf b = load_b_frag(fw1t + (nt * 16 + m) * RBF_DIM + kh * 16);
    acc[nt] = wmma_bf16(a1, b, z);
  }

  // --- bias + ssp, D-layout -> A-layout transpose via LDS (bf16)
#pragma unroll
  for (int nt = 0; nt < 8; ++nt) {
    const int col  = nt * 16 + m;
    const float b1 = fb1[col];
#pragma unroll
    for (int v = 0; v < 8; ++v)
      lds[wave][(v + 8 * kh) * HIDDEN + col] = (__bf16)ssp(acc[nt][v] + b1);
  }
  __syncthreads();

  // --- GEMM2: hidden[16x128] @ fw2[128x128], K in 4 chunks of 32
  v8f w[8];
#pragma unroll
  for (int nt = 0; nt < 8; ++nt) w[nt] = z;
#pragma unroll
  for (int kc = 0; kc < 4; ++kc) {
    Bf16x16 t;
    t.lo = *(const v8bf*)&lds[wave][m * HIDDEN + kc * 32 + kh * 16];
    t.hi = *(const v8bf*)&lds[wave][m * HIDDEN + kc * 32 + kh * 16 + 8];
    const v16bf a2 = __builtin_bit_cast(v16bf, t);
#pragma unroll
    for (int nt = 0; nt < 8; ++nt) {
      const v16bf b = load_b_frag(fw2t + (nt * 16 + m) * HIDDEN + kc * 32 + kh * 16);
      w[nt] = wmma_bf16(a2, b, w[nt]);
    }
  }

  // --- m_ij = h[j] * (W + fb2); scatter-add to aggr[i]
  int iidx[8], jidx[8];
#pragma unroll
  for (int v = 0; v < 8; ++v) {
    iidx[v] = ei[base + 8 * kh + v];
    jidx[v] = ei[N_EDGES + base + 8 * kh + v];
  }
#pragma unroll
  for (int nt = 0; nt < 8; ++nt) {
    const int col  = nt * 16 + m;
    const float b2 = fb2[col];
#pragma unroll
    for (int v = 0; v < 8; ++v) {
      const float val = (w[nt][v] + b2) * h[jidx[v] * HIDDEN + col];
      atomicAdd(&aggr[iidx[v] * HIDDEN + col], val);
    }
  }
  if (lane < 16) atomicAdd(&deg[i0], 1.0f);
}

// ---------------------------------------------------------------------------
// Node kernel: one wave = 16 nodes. normalize -> WMMA GEMM -> ssp -> WMMA GEMM
__global__ __launch_bounds__(256) void node_kernel(
    const float* __restrict__ aggr, const float* __restrict__ deg,
    const __bf16* __restrict__ uw1t,
    const __bf16* __restrict__ uw2t,
    const float* __restrict__ ub1, const float* __restrict__ ub2,
    float* __restrict__ out) {
  __shared__ __bf16 lds[8][16 * HIDDEN];
  const int lane = threadIdx.x & 31;
  const int wave = threadIdx.x >> 5;
  const int m    = lane & 15;
  const int kh   = lane >> 4;
  const int tile = blockIdx.x * 8 + wave;
  if (tile >= N_NODES / 16) return;    // wave-uniform guard (EXEC stays full)

  const int node  = tile * 16 + m;
  const float dv  = deg[node];
  const float inv = 1.0f / fmaxf(dv, 1.0f);
  const v8f z = {0.f, 0.f, 0.f, 0.f, 0.f, 0.f, 0.f, 0.f};

  // --- GEMM1: (aggr/deg)[16x128] @ uw1[128x128]
  v8f acc[8];
#pragma unroll
  for (int nt = 0; nt < 8; ++nt) acc[nt] = z;
#pragma unroll
  for (int kc = 0; kc < 4; ++kc) {
    const float* ap = aggr + (size_t)node * HIDDEN + kc * 32;
    v16bf a;
#pragma unroll
    for (int e = 0; e < 16; ++e) {
      const int k = e + kh * 16;
      a[e] = (__bf16)(ap[k] * inv);
    }
#pragma unroll
    for (int nt = 0; nt < 8; ++nt) {
      const v16bf b = load_b_frag(uw1t + (nt * 16 + m) * HIDDEN + kc * 32 + kh * 16);
      acc[nt] = wmma_bf16(a, b, acc[nt]);
    }
  }

  // --- bias + ssp, transpose via LDS
#pragma unroll
  for (int nt = 0; nt < 8; ++nt) {
    const int col  = nt * 16 + m;
    const float b1 = ub1[col];
#pragma unroll
    for (int v = 0; v < 8; ++v)
      lds[wave][(v + 8 * kh) * HIDDEN + col] = (__bf16)ssp(acc[nt][v] + b1);
  }
  __syncthreads();

  // --- GEMM2: @ uw2[128x128] + ub2 -> h_new
  v8f w[8];
#pragma unroll
  for (int nt = 0; nt < 8; ++nt) w[nt] = z;
#pragma unroll
  for (int kc = 0; kc < 4; ++kc) {
    Bf16x16 t;
    t.lo = *(const v8bf*)&lds[wave][m * HIDDEN + kc * 32 + kh * 16];
    t.hi = *(const v8bf*)&lds[wave][m * HIDDEN + kc * 32 + kh * 16 + 8];
    const v16bf a2 = __builtin_bit_cast(v16bf, t);
#pragma unroll
    for (int nt = 0; nt < 8; ++nt) {
      const v16bf b = load_b_frag(uw2t + (nt * 16 + m) * HIDDEN + kc * 32 + kh * 16);
      w[nt] = wmma_bf16(a2, b, w[nt]);
    }
  }
#pragma unroll
  for (int nt = 0; nt < 8; ++nt) {
    const int col  = nt * 16 + m;
    const float b2 = ub2[col];
#pragma unroll
    for (int v = 0; v < 8; ++v) {
      const int row = tile * 16 + v + 8 * kh;
      out[(size_t)row * HIDDEN + col] = w[nt][v] + b2;
    }
  }
}

// ---------------------------------------------------------------------------
extern "C" void kernel_launch(void* const* d_in, const int* in_sizes, int n_in,
                              void* d_out, int out_size, void* d_ws, size_t ws_size,
                              hipStream_t stream) {
  const float* h   = (const float*)d_in[0];
  const float* x   = (const float*)d_in[1];
  const int*   ei  = (const int*)d_in[2];
  const float* fw1 = (const float*)d_in[3];
  const float* fb1 = (const float*)d_in[4];
  const float* fw2 = (const float*)d_in[5];
  const float* fb2 = (const float*)d_in[6];
  const float* uw1 = (const float*)d_in[7];
  const float* ub1 = (const float*)d_in[8];
  const float* uw2 = (const float*)d_in[9];
  const float* ub2 = (const float*)d_in[10];

  char* ws = (char*)d_ws;
  __bf16* fw1t = (__bf16*)(ws);                                 //  8 KB
  __bf16* fw2t = (__bf16*)(ws + 8192);                          // 32 KB
  __bf16* uw1t = (__bf16*)(ws + 40960);                         // 32 KB
  __bf16* uw2t = (__bf16*)(ws + 73728);                         // 32 KB
  float* aggr = (float*)(ws + 106496);                          // 10.24 MB
  float* deg  = aggr + (size_t)N_NODES * HIDDEN;                // 80 KB (contiguous)
  float* out  = (float*)d_out;

  const int nzero = N_NODES * HIDDEN + N_NODES;                 // aggr + deg
  zero_kernel<<<(nzero + 255) / 256, 256, 0, stream>>>(aggr, nzero);
  prep_kernel<<<(4096 + 3 * 16384 + 255) / 256, 256, 0, stream>>>(
      fw1, fw2, uw1, uw2, fw1t, fw2t, uw1t, uw2t);

  edge_kernel<<<N_EDGES / (16 * 8), 256, 0, stream>>>(
      h, x, ei, fb1, fb2, fw1t, fw2t, aggr, deg);

  const int ntiles = N_NODES / 16;                              // 1250
  node_kernel<<<(ntiles + 7) / 8, 256, 0, stream>>>(
      aggr, deg, uw1t, uw2t, ub1, ub2, out);

  // passthrough: x -> tail of output tuple
  hipMemcpyAsync(out + (size_t)N_NODES * HIDDEN, x,
                 (size_t)N_NODES * 3 * sizeof(float),
                 hipMemcpyDeviceToDevice, stream);
}
